// JBUStack_84224308675052
// MI455X (gfx1250) — matrix operand order repaired and checked by reading, not compiled, not run on hardware
//
#include <hip/hip_runtime.h>
#include <hip/hip_bf16.h>

typedef __bf16  v16bf __attribute__((ext_vector_type(16)));
typedef __bf16  v8bf  __attribute__((ext_vector_type(8)));
typedef __bf16  v2bf  __attribute__((ext_vector_type(2)));
typedef float   v8f   __attribute__((ext_vector_type(8)));

#define RADIUS 3
#define DIAM   7
#define NTAP   49
#define KEYD   32

__device__ __forceinline__ int reflect_idx(int i, int n) {
  if (i < 0) i = -i;
  if (i >= n) i = 2 * n - 2 - i;
  return i;
}
__device__ __forceinline__ int clamp_idx(int i, int n) {
  return i < 0 ? 0 : (i >= n ? n - 1 : i);
}

// ---------------------------------------------------------------------------
// WMMA GEMM, both operands bf16 and K-contiguous:
//   A  : [M][K] bf16 row-major (weights)
//   Bt : [N][K] bf16 row-major (activations, transposed)  -> C = A * Bt^T
// One wave per block; 32(M) x 64(N) tile = 8 WMMAs per K-step.
// A fragment: two contiguous v8bf loads. B fragment: ONE contiguous v16bf load
// (16 contiguous bf16 in K, k = 16*hi + s, same order as the A fragment). No cvt in the loop.
// mode 0: f32 [M][N]        mode 1: bf16T [N][M] with residual R + 0.1*acc
// mode 2: f32 [M][N] + bias
// ---------------------------------------------------------------------------
__global__ __launch_bounds__(32)
void gemm_bf16_nt(const __bf16* __restrict__ A, long aStride,
                  const __bf16* __restrict__ Bt, long bStride,
                  int M, int N, int K, int mode,
                  float* __restrict__ Cf, long cStride,
                  __bf16* __restrict__ CT, long ctStride,
                  const __bf16* __restrict__ RT, long rStride,
                  const float* __restrict__ bias, int biasStride) {
  const int b    = blockIdx.z;
  const int m0   = blockIdx.y * 32;
  const int n0   = blockIdx.x * 64;
  const int lane = threadIdx.x;
  const int row  = lane & 15;
  const int hi   = lane >> 4;

  const __bf16* Ab  = A  + (long)b * aStride;
  const __bf16* Btb = Bt + (long)b * bStride;

  v8f acc[2][4] = {};
  for (int k0 = 0; k0 < K; k0 += 32) {
    v16bf af[2];
#pragma unroll
    for (int m = 0; m < 2; ++m) {
      // A 16x32 frag: lanes<16 hold K{0..15}, lanes>=16 K{16..31}
      const __bf16* Ap = Ab + (long)(m0 + m * 16 + row) * K + k0 + hi * 16;
      v8bf lo = *(const v8bf*)(Ap);
      v8bf hc = *(const v8bf*)(Ap + 8);
      af[m] = __builtin_shufflevector(lo, hc, 0, 1, 2, 3, 4, 5, 6, 7,
                                      8, 9, 10, 11, 12, 13, 14, 15);
    }
#pragma unroll
    for (int t = 0; t < 4; ++t) {
      // B 32x16 frag: lane = column, lanes>=16 take K offset +16; K-contiguous.
      const __bf16* Bp = Btb + (long)(n0 + t * 16 + row) * K + k0 + hi * 16;
      v16bf bv = *(const v16bf*)Bp;
      __builtin_prefetch(Bp + 32, 0, 1);
#pragma unroll
      for (int m = 0; m < 2; ++m)
        acc[m][t] = __builtin_amdgcn_wmma_f32_16x16x32_bf16(
            false, af[m], false, bv, (short)0, acc[m][t], false, false);
    }
  }

#pragma unroll
  for (int t = 0; t < 4; ++t) {
    const int col = n0 + t * 16 + row;
#pragma unroll
    for (int m = 0; m < 2; ++m) {
      const int obase = m0 + m * 16 + 8 * hi;  // rows obase..obase+7 contiguous
      if (mode == 1) {
        const __bf16* rp = RT + (long)b * rStride + (long)col * M + obase;
        v8bf rv = *(const v8bf*)rp;
        v8bf ov;
#pragma unroll
        for (int r = 0; r < 8; ++r)
          ov[r] = (__bf16)((float)rv[r] + 0.1f * acc[m][t][r]);
        *(v8bf*)(CT + (long)b * ctStride + (long)col * M + obase) = ov;
      } else {
#pragma unroll
        for (int r = 0; r < 8; ++r) {
          float v = acc[m][t][r];
          if (mode == 2) v += bias[b * biasStride + obase + r];
          Cf[(long)b * cStride + (long)(obase + r) * N + col] = v;
        }
      }
    }
  }
}

// f32 -> bf16 flat convert (weights).
__global__ void cvt_bf16(const float* __restrict__ in, __bf16* __restrict__ out, long n) {
  long i = (long)blockIdx.x * blockDim.x + threadIdx.x;
  if (i < n) out[i] = (__bf16)in[i];
}

// [B][C][N] f32 -> [B][N][C] bf16 transpose-convert (GEMM B operand prep).
__global__ void tconv_nc(const float* __restrict__ in, __bf16* __restrict__ out,
                         int C, long N) {
  long p = (long)blockIdx.x * blockDim.x + threadIdx.x;
  int b = blockIdx.y;
  if (p >= N) return;
  const float* ip = in + (long)b * C * N + p;
  __bf16* op = out + ((long)b * N + p) * C;
  for (int c = 0; c < C; c += 2) {
    v2bf t;
    t[0] = (__bf16)ip[(long)c * N];
    t[1] = (__bf16)ip[(long)(c + 1) * N];
    *(v2bf*)(op + c) = t;
  }
}

// ---------------------------------------------------------------------------
// Guidance resize (antialiased triangle, factor invScale) + range projection.
// ---------------------------------------------------------------------------
__global__ void guidance_rangeproj(const float* __restrict__ g,
                                   const float* __restrict__ w1, const float* __restrict__ b1,
                                   const float* __restrict__ w2, const float* __restrict__ b2,
                                   float* __restrict__ outp,
                                   int H, int W, float invScale) {
  const int GH = 768, GW = 1280;
  int idx = blockIdx.x * blockDim.x + threadIdx.x;
  int total = 2 * H * W;
  if (idx >= total) return;
  int x = idx % W;
  int y = (idx / W) % H;
  int b = idx / (W * H);

  float cy = (y + 0.5f) * invScale - 0.5f;
  float cx = (x + 0.5f) * invScale - 0.5f;
  float r  = invScale;
  int y0 = (int)ceilf(cy - r), y1 = (int)floorf(cy + r);
  int x0 = (int)ceilf(cx - r), x1 = (int)floorf(cx + r);

  float rgb0 = 0.f, rgb1 = 0.f, rgb2 = 0.f, wsum = 0.f;
  const float* gb = g + (long)b * 3 * GH * GW;
  for (int iy = y0; iy <= y1; ++iy) {
    float wy = 1.0f - fabsf((float)iy - cy) / r;
    if (wy <= 0.f) continue;
    int ry = clamp_idx(iy, GH);
    for (int ix = x0; ix <= x1; ++ix) {
      float wx = 1.0f - fabsf((float)ix - cx) / r;
      if (wx <= 0.f) continue;
      int rx = clamp_idx(ix, GW);
      float w = wy * wx;
      long base = (long)ry * GW + rx;
      wsum += w;
      rgb0 = fmaf(w, gb[base], rgb0);
      rgb1 = fmaf(w, gb[(long)GH * GW + base], rgb1);
      rgb2 = fmaf(w, gb[2l * GH * GW + base], rgb2);
    }
  }
  float inv = 1.0f / wsum;
  rgb0 *= inv; rgb1 *= inv; rgb2 *= inv;

  float hbuf[KEYD];
#pragma unroll
  for (int o = 0; o < KEYD; ++o) {
    float a = b1[o] + w1[o * 3 + 0] * rgb0 + w1[o * 3 + 1] * rgb1 + w1[o * 3 + 2] * rgb2;
    hbuf[o] = 0.5f * a * (1.0f + erff(a * 0.70710678118654752f));
  }
  float* op = outp + (long)idx * KEYD;
#pragma unroll 4
  for (int k = 0; k < KEYD; ++k) {
    float a = b2[k];
#pragma unroll
    for (int o = 0; o < KEYD; ++o) a = fmaf(w2[k * KEYD + o], hbuf[o], a);
    op[k] = a;
  }
}

// 2x bicubic upsample (a=-0.75, half-pixel, edge clamp); bf16 output.
__global__ void bicubic_up2x(const float* __restrict__ in, __bf16* __restrict__ out,
                             int C, int h, int w) {
  const int H = 2 * h, W = 2 * w;
  long idx = (long)blockIdx.x * blockDim.x + threadIdx.x;
  long total = 2l * C * H * W;
  if (idx >= total) return;
  int X = (int)(idx % W);
  long r1 = idx / W;
  int Y = (int)(r1 % H);
  long r2 = r1 / H;
  int c = (int)(r2 % C);
  int b = (int)(r2 / C);
  int iy = Y >> 1, ix = X >> 1, ey = Y & 1, ex = X & 1;
  const float WEv[4] = {-0.03515625f, 0.26171875f, 0.87890625f, -0.10546875f};
  const float WOv[4] = {-0.10546875f, 0.87890625f, 0.26171875f, -0.03515625f};
  const float* ip = in + (long)(b * C + c) * h * w;
  float acc = 0.f;
#pragma unroll
  for (int jy = 0; jy < 4; ++jy) {
    float wy = ey ? WOv[jy] : WEv[jy];
    int sy = clamp_idx(iy + jy - (ey ? 1 : 2), h);
    float rowv = 0.f;
#pragma unroll
    for (int jx = 0; jx < 4; ++jx) {
      float wx = ex ? WOv[jx] : WEv[jx];
      int sx = clamp_idx(ix + jx - (ex ? 1 : 2), w);
      rowv = fmaf(wx, ip[(long)sy * w + sx], rowv);
    }
    acc = fmaf(wy, rowv, acc);
  }
  out[idx] = (__bf16)acc;
}

// ---------------------------------------------------------------------------
// JBU apply. hr is bf16. outMode 0: f32 [C][H][W]; outMode 1: bf16 [H*W][C]
// (v8bf store per 8-channel chunk -> direct GEMM-B format for the next stage).
// ---------------------------------------------------------------------------
__global__ __launch_bounds__(256)
void jbu_apply(const float* __restrict__ proj, const __bf16* __restrict__ hr,
               const float* __restrict__ temp_p, const float* __restrict__ sigma_p,
               float* __restrict__ outF, __bf16* __restrict__ outT, int outMode,
               int C, int H, int W) {
  __shared__ float sbuf[22 * 22 * KEYD];  // 61,952 B; reused as 22*22*8 in phase 2
  const int b   = blockIdx.z;
  const int ty0 = blockIdx.y * 16, tx0 = blockIdx.x * 16;
  const int tid = threadIdx.y * 16 + threadIdx.x;

  for (int i = tid; i < 22 * 22 * KEYD; i += 256) {
    int k = i & 31, rem = i >> 5;
    int xx = rem % 22, yy = rem / 22;
    int gy = reflect_idx(ty0 - RADIUS + yy, H);
    int gx = reflect_idx(tx0 - RADIUS + xx, W);
    sbuf[i] = proj[(((long)b * H + gy) * W + gx) * KEYD + k];
  }
  __syncthreads();

  const float* own = &sbuf[((threadIdx.y + RADIUS) * 22 + threadIdx.x + RADIUS) * KEYD];
  float tclip = expf(temp_p[0]);
  tclip = fminf(fmaxf(tclip, 1e-4f), 1e4f);
  float sigma = sigma_p[0];
  float sinv  = -1.0f / (2.0f * sigma * sigma);

  float ck[NTAP];
  float mx = -3.4e38f;
#pragma unroll
  for (int s = 0; s < NTAP; ++s) {
    int dy = s / DIAM, dx = s % DIAM;
    const float* nb = &sbuf[((threadIdx.y + dy) * 22 + threadIdx.x + dx) * KEYD];
    float dot = 0.f;
    for (int k = 0; k < KEYD; ++k) dot = fmaf(own[k], nb[k], dot);
    dot *= tclip;
    ck[s] = dot;
    mx = fmaxf(mx, dot);
  }
  float sum = 0.f;
#pragma unroll
  for (int s = 0; s < NTAP; ++s) {
    int dy = s / DIAM, dx = s % DIAM;
    float dyd = (float)(dy - RADIUS) * (1.0f / RADIUS);
    float dxd = (float)(dx - RADIUS) * (1.0f / RADIUS);
    float sp = expf((dyd * dyd + dxd * dxd) * sinv);
    float e = expf(ck[s] - mx) * sp;
    ck[s] = e;
    sum += e;
  }
  float rsum = 1.0f / fmaxf(sum, 1e-30f);
#pragma unroll
  for (int s = 0; s < NTAP; ++s) ck[s] *= rsum;
  __syncthreads();

  const int py = ty0 + threadIdx.y, px = tx0 + threadIdx.x;
  for (int c0 = 0; c0 < C; c0 += 8) {
    for (int i = tid; i < 22 * 22 * 8; i += 256) {
      int c = i & 7, rem = i >> 3;
      int xx = rem % 22, yy = rem / 22;
      int gy = reflect_idx(ty0 - RADIUS + yy, H);
      int gx = reflect_idx(tx0 - RADIUS + xx, W);
      sbuf[i] = (float)hr[(((long)b * C + c0 + c) * H + gy) * W + gx];
    }
    __syncthreads();
    float acc[8] = {};
#pragma unroll
    for (int s = 0; s < NTAP; ++s) {
      int dy = s / DIAM, dx = s % DIAM;
      const float* nb = &sbuf[((threadIdx.y + dy) * 22 + threadIdx.x + dx) * 8];
      float wgt = ck[s];
#pragma unroll
      for (int c = 0; c < 8; ++c) acc[c] = fmaf(wgt, nb[c], acc[c]);
    }
    if (outMode == 0) {
#pragma unroll
      for (int c = 0; c < 8; ++c)
        outF[(((long)b * C + c0 + c) * H + py) * W + px] = acc[c];
    } else {
      v8bf ov;
#pragma unroll
      for (int c = 0; c < 8; ++c) ov[c] = (__bf16)acc[c];
      *(v8bf*)(outT + ((long)b * H * W + (long)py * W + px) * C + c0) = ov;
    }
    __syncthreads();
  }
}

// ---------------------------------------------------------------------------
// GroupNorm over the bf16 transposed tensor: partial sums (coalesced over C)
// + atomics, then finalize into per-channel scale/shift for GEMM folding.
// ---------------------------------------------------------------------------
__global__ void zero_f32(float* __restrict__ p, int n) {
  int i = blockIdx.x * blockDim.x + threadIdx.x;
  if (i < n) p[i] = 0.f;
}

__global__ __launch_bounds__(256)
void gn_partial(const __bf16* __restrict__ xT, float* __restrict__ accum,
                int C, long N, int chunks) {
  int b = blockIdx.y, chunk = blockIdx.x, c = threadIdx.x;
  long p0 = (long)chunk * N / chunks, p1 = (long)(chunk + 1) * N / chunks;
  float s = 0.f, q = 0.f;
  for (long p = p0; p < p1; ++p) {
    float v = (float)xT[((long)b * N + p) * C + c];
    s += v;
    q = fmaf(v, v, q);
  }
  atomicAdd(&accum[(b * C + c) * 2 + 0], s);
  atomicAdd(&accum[(b * C + c) * 2 + 1], q);
}

__global__ void gn_finalize(const float* __restrict__ accum,
                            const float* __restrict__ gw, const float* __restrict__ gb,
                            float* __restrict__ sc, float* __restrict__ sh,
                            int C, long HW) {
  int b = blockIdx.x, c = threadIdx.x;
  int g0 = (c / 8) * 8;
  float s = 0.f, q = 0.f;
  for (int j = 0; j < 8; ++j) {
    s += accum[(b * C + g0 + j) * 2 + 0];
    q += accum[(b * C + g0 + j) * 2 + 1];
  }
  float n = (float)(HW * 8);
  float mean = s / n;
  float var = q / n - mean * mean;
  float rstd = rsqrtf(var + 1e-5f);
  float scale = gw[c] * rstd;
  sc[b * C + c] = scale;
  sh[b * C + c] = gb[c] - mean * scale;
}

// Fold GN scale/shift into per-batch bf16 projection weights + f32 bias.
__global__ __launch_bounds__(256)
void fold_proj(const float* __restrict__ pw, const float* __restrict__ sc,
               const float* __restrict__ sh, __bf16* __restrict__ PwB,
               float* __restrict__ biasv, int C) {
  int b = blockIdx.x, o = threadIdx.x;
  float acc = 0.f;
  for (int c = 0; c < C; ++c) {
    float w = pw[o * C + c];
    PwB[((long)b * C + o) * C + c] = (__bf16)(w * sc[b * C + c]);
    acc = fmaf(w, sh[b * C + c], acc);
  }
  biasv[b * C + o] = acc;
}

// ---------------------------------------------------------------------------
extern "C" void kernel_launch(void* const* d_in, const int* in_sizes, int n_in,
                              void* d_out, int out_size, void* d_ws, size_t ws_size,
                              hipStream_t stream) {
  (void)in_sizes; (void)n_in; (void)out_size; (void)ws_size;
  const float* source   = (const float*)d_in[0];
  const float* guidance = (const float*)d_in[1];
  const float* pre_w    = (const float*)d_in[2];
  const float* rp1_w1   = (const float*)d_in[3];
  const float* rp1_b1   = (const float*)d_in[4];
  const float* rp1_w2   = (const float*)d_in[5];
  const float* rp1_b2   = (const float*)d_in[6];
  const float* temp1    = (const float*)d_in[7];
  const float* sigma1   = (const float*)d_in[8];
  const float* rp2_w1   = (const float*)d_in[9];
  const float* rp2_b1   = (const float*)d_in[10];
  const float* rp2_w2   = (const float*)d_in[11];
  const float* rp2_b2   = (const float*)d_in[12];
  const float* temp2    = (const float*)d_in[13];
  const float* sigma2   = (const float*)d_in[14];
  const float* fixup_w  = (const float*)d_in[15];
  const float* gn_w     = (const float*)d_in[16];
  const float* gn_b     = (const float*)d_in[17];
  const float* proj_w   = (const float*)d_in[18];

  const int MD = 256, FD = 384;
  const int h0 = 48, w0 = 80;
  const int H1 = 96, W1 = 160, H2 = 192, W2 = 320;
  const long N0 = (long)h0 * w0;   // 3840
  const long N1 = (long)H1 * W1;   // 15360
  const long N2 = (long)H2 * W2;   // 61440

  // Workspace carve-up (bytes).
  char* ws = (char*)d_ws;
  size_t off = 0;
  auto carve = [&](size_t bytes) { void* p = ws + off; off += (bytes + 255) & ~(size_t)255; return p; };
  __bf16* srcT  = (__bf16*)carve(2 * N0 * FD * 2);        // [2][3840][384]
  __bf16* wpreB = (__bf16*)carve((size_t)MD * FD * 2);    // [256][384]
  __bf16* wfixB = (__bf16*)carve((size_t)MD * MD * 2);    // [256][256]
  float*  x0    = (float*) carve(2 * (size_t)MD * N0 * 4);// [2][256][48][80]
  float*  proj1 = (float*) carve(2 * N1 * KEYD * 4);      // [2][96][160][32]
  __bf16* hr1   = (__bf16*)carve(2 * (size_t)MD * N1 * 2);// [2][256][96][160]
  float*  x1    = (float*) carve(2 * (size_t)MD * N1 * 4);// [2][256][96][160]
  float*  proj2 = (float*) carve(2 * N2 * KEYD * 4);      // [2][192][320][32]
  __bf16* hr2   = (__bf16*)carve(2 * (size_t)MD * N2 * 2);// [2][256][192][320]
  __bf16* xT2   = (__bf16*)carve(2 * N2 * (size_t)MD * 2);// [2][61440][256]
  __bf16* xfT   = (__bf16*)carve(2 * N2 * (size_t)MD * 2);// [2][61440][256]
  float*  accum = (float*) carve(2 * MD * 2 * 4);
  float*  scv   = (float*) carve(2 * MD * 4);
  float*  shv   = (float*) carve(2 * MD * 4);
  __bf16* PwB   = (__bf16*)carve(2 * (size_t)MD * MD * 2);
  float*  biasv = (float*) carve(2 * MD * 4);

  // 0. weight converts + source transpose-convert
  cvt_bf16<<<(unsigned)((MD * FD + 255) / 256), 256, 0, stream>>>(pre_w, wpreB, (long)MD * FD);
  cvt_bf16<<<(unsigned)((MD * MD + 255) / 256), 256, 0, stream>>>(fixup_w, wfixB, (long)MD * MD);
  tconv_nc<<<dim3((unsigned)((N0 + 255) / 256), 2), 256, 0, stream>>>(source, srcT, FD, N0);

  // 1. pre_proj GEMM -> x0 f32 [C][h][w]
  gemm_bf16_nt<<<dim3((unsigned)(N0 / 64), MD / 32, 2), 32, 0, stream>>>(
      wpreB, 0, srcT, N0 * FD, MD, (int)N0, FD, 0,
      x0, (long)MD * N0, nullptr, 0, nullptr, 0, nullptr, 0);

  // 2. level-1 guidance resize (1/8) + range projection
  guidance_rangeproj<<<(2 * H1 * W1 + 127) / 128, 128, 0, stream>>>(
      guidance, rp1_w1, rp1_b1, rp1_w2, rp1_b2, proj1, H1, W1, 8.0f);

  // 3. bicubic x0 -> hr1 (bf16)
  bicubic_up2x<<<(unsigned)((2l * MD * N1 + 255) / 256), 256, 0, stream>>>(x0, hr1, MD, h0, w0);

  // 4. JBU level 1 -> x1 f32 CHW
  jbu_apply<<<dim3(W1 / 16, H1 / 16, 2), dim3(16, 16), 0, stream>>>(
      proj1, hr1, temp1, sigma1, x1, nullptr, 0, MD, H1, W1);

  // 5. level-2 guidance resize (1/4) + range projection
  guidance_rangeproj<<<(2 * H2 * W2 + 127) / 128, 128, 0, stream>>>(
      guidance, rp2_w1, rp2_b1, rp2_w2, rp2_b2, proj2, H2, W2, 4.0f);

  // 6. bicubic x1 -> hr2 (bf16)
  bicubic_up2x<<<(unsigned)((2l * MD * N2 + 255) / 256), 256, 0, stream>>>(x1, hr2, MD, H1, W1);

  // 7. JBU level 2 -> xT2 bf16 [pix][C] (GEMM-ready)
  jbu_apply<<<dim3(W2 / 16, H2 / 16, 2), dim3(16, 16), 0, stream>>>(
      proj2, hr2, temp2, sigma2, nullptr, xT2, 1, MD, H2, W2);

  // 8. FixUp GEMM: xfT = xT2 + 0.1 * (fixup_w @ x2), bf16T output
  gemm_bf16_nt<<<dim3((unsigned)(N2 / 64), MD / 32, 2), 32, 0, stream>>>(
      wfixB, 0, xT2, N2 * (long)MD, MD, (int)N2, MD, 1,
      nullptr, 0, xfT, N2 * (long)MD, xT2, N2 * (long)MD, nullptr, 0);

  // 9. GroupNorm stats (coalesced over C) -> per-channel scale/shift
  zero_f32<<<4, 256, 0, stream>>>(accum, 2 * MD * 2);
  gn_partial<<<dim3(64, 2), 256, 0, stream>>>(xfT, accum, MD, N2, 64);
  gn_finalize<<<2, 256, 0, stream>>>(accum, gn_w, gn_b, scv, shv, MD, N2);

  // 10. Fold GN into bf16 projection weights
  fold_proj<<<2, 256, 0, stream>>>(proj_w, scv, shv, PwB, biasv, MD);

  // 11. Final projection GEMM: d_out = Pw[b] @ xf + bias[b], f32 [C][H][W]
  gemm_bf16_nt<<<dim3((unsigned)(N2 / 64), MD / 32, 2), 32, 0, stream>>>(
      PwB, (long)MD * MD, xfT, N2 * (long)MD, MD, (int)N2, MD, 2,
      (float*)d_out, (long)MD * N2, nullptr, 0, nullptr, 0, biasv, MD);
}
